// GNN_65824668779033
// MI455X (gfx1250) — matrix operand order repaired and checked by reading, not compiled, not run on hardware
//
#include <hip/hip_runtime.h>
#include <hip/hip_bf16.h>

// Problem constants (match the reference)
#define NL   200000      // literals
#define NCL  800000      // clauses
#define NE   2400000     // edges
#define NVAR (NL / 2)

typedef __attribute__((ext_vector_type(16))) __bf16 v16bf;
typedef __attribute__((ext_vector_type(8)))  float  v8f;

union AF { uint4 q[2]; v16bf v; };

__device__ __forceinline__ unsigned short f2bf(float f) {
  unsigned u = __float_as_uint(f);
  u += 0x7FFFu + ((u >> 16) & 1u);          // round to nearest even
  return (unsigned short)(u >> 16);
}
__device__ __forceinline__ float bf2f(unsigned hi) { return __uint_as_float(hi << 16); }
__device__ __forceinline__ float silu_f(float x) { return x / (1.0f + __expf(-x)); }

// ---------------------------------------------------------------------------
// Encoder stage 1: t = silu(x @ W1 + b1), x:[N,4] f32, W1:[4,64], t:[N,64] bf16
// one thread per (row, channel-pair)
// ---------------------------------------------------------------------------
__global__ void sat_enc1(const float* __restrict__ x, const float* __restrict__ w1,
                         const float* __restrict__ b1, unsigned short* __restrict__ t,
                         long N) {
  long idx = (long)blockIdx.x * blockDim.x + threadIdx.x;
  long row = idx >> 5;
  int  p   = (int)(idx & 31);
  if (row >= N) return;
  const float* xr = x + row * 4;
  float x0 = xr[0], x1 = xr[1], x2 = xr[2], x3 = xr[3];
  int c = 2 * p;
  float a0 = b1[c]     + x0 * w1[c]     + x1 * w1[64 + c]     + x2 * w1[128 + c]     + x3 * w1[192 + c];
  float a1 = b1[c + 1] + x0 * w1[c + 1] + x1 * w1[64 + c + 1] + x2 * w1[128 + c + 1] + x3 * w1[192 + c + 1];
  a0 = silu_f(a0);
  a1 = silu_f(a1);
  unsigned pk = (unsigned)f2bf(a0) | ((unsigned)f2bf(a1) << 16);
  *(unsigned*)(t + row * 64 + c) = pk;
}

// ---------------------------------------------------------------------------
// Edge degree counts (once; degrees are layer-invariant)
// ---------------------------------------------------------------------------
__global__ void sat_count(const int* __restrict__ el, const int* __restrict__ ec,
                          float* __restrict__ cl, float* __restrict__ cc, int E) {
  int e = blockIdx.x * blockDim.x + threadIdx.x;
  if (e >= E) return;
  atomicAdd(&cl[el[e]], 1.0f);
  atomicAdd(&cc[ec[e]], 1.0f);
}

__global__ void sat_inv(float* __restrict__ c, int N) {
  int i = blockIdx.x * blockDim.x + threadIdx.x;
  if (i >= N) return;
  c[i] = 1.0f / fmaxf(c[i], 1.0f);
}

// ---------------------------------------------------------------------------
// Edge scatter-add: one wave per edge (edge id made wave-uniform via
// readfirstlane so index fetches become scalar loads), 2 channels per lane,
// fp32 atomics into the accumulator.
// ---------------------------------------------------------------------------
__global__ void sat_scatter(const unsigned short* __restrict__ h,
                            const int* __restrict__ si, const int* __restrict__ di,
                            float* __restrict__ acc, int E) {
  int wid  = __builtin_amdgcn_readfirstlane((int)(threadIdx.x >> 5));   // SGPR wave id
  int e    = blockIdx.x * 8 + wid;                                      // SGPR edge id
  int lane = (int)(threadIdx.x & 31);
  if (e >= E) return;
  int s = si[e];                                                        // scalar loads
  int d = di[e];
  unsigned pk = *(const unsigned*)(h + (long)s * 64 + lane * 2);
  float* ap = acc + (long)d * 64 + lane * 2;
  atomicAdd(ap,     bf2f(pk & 0xFFFFu));
  atomicAdd(ap + 1, bf2f(pk >> 16));
}

// m_bf16 = acc_f32 * inv_count (per node row)
__global__ void sat_scale(const float* __restrict__ acc, const float* __restrict__ inv,
                          unsigned short* __restrict__ m, long N) {
  long g = (long)blockIdx.x * blockDim.x + threadIdx.x;
  long row = g >> 5;
  int p = (int)(g & 31);
  if (row >= N) return;
  float iv = inv[row];
  const float* ap = acc + row * 64 + 2 * p;
  unsigned pk = (unsigned)f2bf(ap[0] * iv) | ((unsigned)f2bf(ap[1] * iv) << 16);
  *(unsigned*)(m + row * 64 + 2 * p) = pk;
}

// ---------------------------------------------------------------------------
// Generalized WMMA GEMM (K and activation are compile-time -> fully unrolled
// K loop, software-pipelinable loads, constant-folded source selection):
//   out[row, c0+j] = act( sum_k A[row,k] * W[k, c0+j] + bias[c0+j] )
// A is the implicit concat of up to 3 bf16 sources (64 K each), with per-source
// row transform: r = (row*mul + add) ^ xor  (flip = xor 1, even/odd = mul 2).
// One wave computes a 16x64 tile: 4 x v_wmma_f32_16x16x32_bf16 per K-block.
// Weights (fp32 global) staged transposed+bf16 into LDS.
// N must be a multiple of 16; guards are wave-uniform so EXEC stays all-ones.
// ---------------------------------------------------------------------------
template <int K, int ACT>
__global__ __launch_bounds__(256) void sat_wmma_gemm(
    const unsigned short* __restrict__ s0,
    const unsigned short* __restrict__ s1,
    const unsigned short* __restrict__ s2,
    int mul0, int add0, int xr0,
    int mul1, int add1, int xr1,
    int mul2, int add2, int xr2,
    const float* __restrict__ W, int ldw,
    const float* __restrict__ bias,
    unsigned short* __restrict__ out, int ldo,
    int N) {
  constexpr int KST = K + 8;                       // padded LDS stride (bf16 elems)
  __shared__ __align__(16) unsigned short Wt[64 * KST];
  const int c0 = blockIdx.y * 64;

  // stage weight slice (64 output cols), transposed, bf16
  for (int i = threadIdx.x; i < 64 * K; i += blockDim.x) {
    int col = i / K;
    int k   = i - col * K;
    Wt[col * KST + k] = f2bf(W[(long)k * ldw + c0 + col]);
  }
  __syncthreads();

  const int lane = threadIdx.x & 31;
  const int wave = threadIdx.x >> 5;
  const long rowBase = ((long)blockIdx.x * 8 + wave) * 16;
  if (rowBase >= N) return;                        // wave-uniform exit

  const int  half = lane >> 4;                     // K-half selector
  const int  lm   = lane & 15;                     // row (A) / col (B,D) within tile
  const long aRow = rowBase + lm;

  const v8f vz = {0.f, 0.f, 0.f, 0.f, 0.f, 0.f, 0.f, 0.f};
  v8f acc[4];
  acc[0] = vz; acc[1] = vz; acc[2] = vz; acc[3] = vz;

  constexpr int KB = K >> 5;                       // number of 32-wide K blocks
#pragma unroll
  for (int kb = 0; kb < KB; ++kb) {
    constexpr int dummy = 0; (void)dummy;
    const int sIdx = kb >> 1;                      // constant after unroll
    const unsigned short* sp;
    int mul, add, xr;
    if (sIdx == 0)      { sp = s0; mul = mul0; add = add0; xr = xr0; }
    else if (sIdx == 1) { sp = s1; mul = mul1; add = add1; xr = xr1; }
    else                { sp = s2; mul = mul2; add = add2; xr = xr2; }

    long r = (aRow * mul + add) ^ xr;
    const unsigned short* rp = sp + r * 64 + (kb & 1) * 32;

    // A fragment 16x32 bf16: lane(row=lm, half) holds K {8h..8h+7, 16+8h..16+8h+7}
    AF a;
    a.q[0] = *(const uint4*)(rp + 8 * half);
    a.q[1] = *(const uint4*)(rp + 16 + 8 * half);

#pragma unroll
    for (int t = 0; t < 4; ++t) {
      // B fragment 32x16 bf16: lane(col=lm, half) holds K {8h..8h+7, 16+8h..16+8h+7}
      const unsigned short* wp = &Wt[(16 * t + lm) * KST + kb * 32 + 8 * half];
      AF b;
      b.q[0] = *(const uint4*)(wp);
      b.q[1] = *(const uint4*)(wp + 16);
      acc[t] = __builtin_amdgcn_wmma_f32_16x16x32_bf16(
          false, a.v, false, b.v, (short)0, acc[t], false, false);
    }
  }

  // epilogue: D element (row = rr + 8*half, col = c0 + 16t + lm)
#pragma unroll
  for (int t = 0; t < 4; ++t) {
    float bv = bias[c0 + 16 * t + lm];
#pragma unroll
    for (int rr = 0; rr < 8; ++rr) {
      float v = acc[t][rr] + bv;
      if (ACT) v = silu_f(v);
      long row = rowBase + rr + 8 * half;
      out[row * (long)ldo + c0 + 16 * t + lm] = f2bf(v);
    }
  }
}

// ---------------------------------------------------------------------------
// Final readout: y = t_var[100k,128] @ out_w2[128,2] + out_b2   (fp32 out)
// ---------------------------------------------------------------------------
__global__ void sat_readout2(const unsigned short* __restrict__ tv,
                             const float* __restrict__ w2, const float* __restrict__ b2,
                             float* __restrict__ y, int NV) {
  int v = blockIdx.x * blockDim.x + threadIdx.x;
  if (v >= NV) return;
  float a0 = b2[0], a1 = b2[1];
  const unsigned short* tr = tv + (long)v * 128;
#pragma unroll 4
  for (int k = 0; k < 128; k += 2) {
    unsigned pk = *(const unsigned*)(tr + k);
    float t0 = bf2f(pk & 0xFFFFu), t1 = bf2f(pk >> 16);
    a0 += t0 * w2[2 * k]     + t1 * w2[2 * k + 2];
    a1 += t0 * w2[2 * k + 1] + t1 * w2[2 * k + 3];
  }
  y[2 * v]     = a0;
  y[2 * v + 1] = a1;
}

// ---------------------------------------------------------------------------
extern "C" void kernel_launch(void* const* d_in, const int* in_sizes, int n_in,
                              void* d_out, int out_size, void* d_ws, size_t ws_size,
                              hipStream_t stream) {
  (void)in_sizes; (void)n_in; (void)out_size; (void)ws_size;

  const float* x_lit = (const float*)d_in[0];
  const float* x_cls = (const float*)d_in[1];
  const int*   e_lit = (const int*)d_in[2];
  const int*   e_cls = (const int*)d_in[3];
  const float* ew1   = (const float*)d_in[4];
  const float* eb1   = (const float*)d_in[5];
  const float* ew2   = (const float*)d_in[6];
  const float* eb2   = (const float*)d_in[7];
  const float* Wc    = (const float*)d_in[8];
  const float* bc    = (const float*)d_in[9];
  const float* Wl    = (const float*)d_in[10];
  const float* bl    = (const float*)d_in[11];
  const float* ow1   = (const float*)d_in[12];
  const float* ob1   = (const float*)d_in[13];
  const float* ow2   = (const float*)d_in[14];
  const float* ob2   = (const float*)d_in[15];
  float* y = (float*)d_out;

  // workspace layout
  size_t off = 0;
  auto alloc = [&](size_t bytes) -> char* {
    char* p = (char*)d_ws + off;
    off += (bytes + 255) & ~(size_t)255;
    return p;
  };
  unsigned short* h_lit  = (unsigned short*)alloc((size_t)NL  * 64 * 2);
  unsigned short* h_cls  = (unsigned short*)alloc((size_t)NCL * 64 * 2);
  unsigned short* m_cls  = (unsigned short*)alloc((size_t)NCL * 64 * 2);
  unsigned short* m_lit  = (unsigned short*)alloc((size_t)NL  * 64 * 2);
  float*          accum  = (float*)         alloc((size_t)NCL * 64 * 4);
  float*          cnt_c  = (float*)         alloc((size_t)NCL * 4);
  float*          cnt_l  = (float*)         alloc((size_t)NL  * 4);
  unsigned short* tbuf   = (unsigned short*)accum;  // encoder hidden reuse (bf16)
  unsigned short* tvar   = m_cls;                   // readout hidden reuse

  // --- degree counts -> inverse (layer invariant) ---
  hipMemsetAsync(cnt_c, 0, (size_t)NCL * 4, stream);
  hipMemsetAsync(cnt_l, 0, (size_t)NL  * 4, stream);
  sat_count<<<(NE + 255) / 256, 256, 0, stream>>>(e_lit, e_cls, cnt_l, cnt_c, NE);
  sat_inv<<<(NCL + 255) / 256, 256, 0, stream>>>(cnt_c, NCL);
  sat_inv<<<(NL  + 255) / 256, 256, 0, stream>>>(cnt_l, NL);

  // --- encoder: h = silu(x@W1+b1) @ W2 + b2 ---
  sat_enc1<<<(unsigned)(((long)NL * 32 + 255) / 256), 256, 0, stream>>>(x_lit, ew1, eb1, tbuf, NL);
  sat_wmma_gemm<64, 0><<<dim3((NL + 127) / 128, 1), 256, 0, stream>>>(
      tbuf, tbuf, tbuf, 1, 0, 0, 1, 0, 0, 1, 0, 0,
      ew2, 64, eb2, h_lit, 64, NL);
  sat_enc1<<<(unsigned)(((long)NCL * 32 + 255) / 256), 256, 0, stream>>>(x_cls, ew1, eb1, tbuf, NCL);
  sat_wmma_gemm<64, 0><<<dim3((NCL + 127) / 128, 1), 256, 0, stream>>>(
      tbuf, tbuf, tbuf, 1, 0, 0, 1, 0, 0, 1, 0, 0,
      ew2, 64, eb2, h_cls, 64, NCL);

  // --- message-passing layers ---
  for (int l = 0; l < 4; ++l) {
    // lit -> cls
    hipMemsetAsync(accum, 0, (size_t)NCL * 64 * 4, stream);
    sat_scatter<<<(NE + 7) / 8, 256, 0, stream>>>(h_lit, e_lit, e_cls, accum, NE);
    sat_scale<<<(unsigned)(((long)NCL * 32 + 255) / 256), 256, 0, stream>>>(
        accum, cnt_c, m_cls, NCL);
    // hc = silu([hc | m_cls] @ Wc[l] + bc[l])   (in-place: rows private per wave)
    sat_wmma_gemm<128, 1><<<dim3((NCL + 127) / 128, 1), 256, 0, stream>>>(
        h_cls, m_cls, m_cls, 1, 0, 0, 1, 0, 0, 1, 0, 0,
        Wc + (size_t)l * 128 * 64, 64, bc + l * 64, h_cls, 64, NCL);

    // cls -> lit
    hipMemsetAsync(accum, 0, (size_t)NL * 64 * 4, stream);
    sat_scatter<<<(NE + 7) / 8, 256, 0, stream>>>(h_cls, e_cls, e_lit, accum, NE);
    sat_scale<<<(unsigned)(((long)NL * 32 + 255) / 256), 256, 0, stream>>>(
        accum, cnt_l, m_lit, NL);
    // hl = silu([hl | m_lit | hl_flip] @ Wl[l] + bl[l]); flip = row^1 stays in-tile
    sat_wmma_gemm<192, 1><<<dim3((NL + 127) / 128, 1), 256, 0, stream>>>(
        h_lit, m_lit, h_lit, 1, 0, 0, 1, 0, 0, 1, 0, 1,
        Wl + (size_t)l * 192 * 64, 64, bl + l * 64, h_lit, 64, NL);
  }

  // --- variable readout ---
  // t_var = silu([h_lit[0::2] | h_lit[1::2]] @ out_w1 + out_b1), 128 output cols
  sat_wmma_gemm<128, 1><<<dim3((NVAR + 127) / 128, 2), 256, 0, stream>>>(
      h_lit, h_lit, h_lit, 2, 0, 0, 2, 1, 0, 1, 0, 0,
      ow1, 128, ob1, tvar, 128, NVAR);
  sat_readout2<<<(NVAR + 255) / 256, 256, 0, stream>>>(tvar, ow2, ob2, y, NVAR);
}
